// TriangleMultiplication_40819369181353
// MI455X (gfx1250) — matrix operand order repaired and checked by reading, not compiled, not run on hardware
//
#include <hip/hip_runtime.h>
#include <hip/hip_bf16.h>

// ---------------------------------------------------------------------------
// Triangle multiplication ("outgoing") for B=1, N=512, C=128, fp32 reference.
// f16 WMMA (v_wmma_f32_16x16x32_f16) for every matmul; CDNA5 data movers:
//   k1: X tile loaded to LDS via TDM tensor_load_to_lds (+ s_wait_tensorcnt)
//   k2: A/B tiles double-buffered into LDS via global_load_async_to_lds_b128
//       (ASYNCcnt, in-order completion -> s_wait_asynccnt 12 overlaps the
//       next DMA stage with WMMA on the current stage).
// Pipeline:
//   k0: fp32 -> f16 weight conversion (6 x 128x128) into ws
//   k1: fused LN1 + left/right MLPs + gate; A,B written f16 channel-major
//       (c,i,k) so the triangle einsum becomes 128 plain A*B^T GEMMs.
//   k2: per-channel 512x512x512 GEMM, T (i,j,c) fp32 into d_out (scratch).
//   k3: LN2 + output linear (WMMA) * gate, in place on d_out.
// ---------------------------------------------------------------------------

#define NPOS 512
#define CDIM 128
#define NN   (NPOS * NPOS)

typedef __attribute__((ext_vector_type(16))) _Float16 v16h;
typedef __attribute__((ext_vector_type(8)))  _Float16 v8h;
typedef __attribute__((ext_vector_type(2)))  _Float16 h2;
typedef __attribute__((ext_vector_type(8)))  float    v8f;
typedef __attribute__((ext_vector_type(4)))  unsigned u32x4;
typedef __attribute__((ext_vector_type(8)))  unsigned u32x8;

// workspace layout (bytes)
#define OFF_W16 0                                   // 6 * 128*128 f16
#define OFF_AT  (262144)                            // 128 * 512*512 f16 (64MB)
#define OFF_BT  (OFF_AT + (size_t)NN * CDIM * 2)    // 64MB
#define OFF_G   (OFF_BT + (size_t)NN * CDIM * 2)    // 128MB fp32

// weight slots inside W16
#define MAT_LW1 0
#define MAT_LW2 1
#define MAT_RW1 2
#define MAT_RW2 3
#define MAT_SW  4
#define MAT_CW  5

__device__ __forceinline__ float sigmoidf_(float x) {
    return 1.0f / (1.0f + __expf(-x));
}

// low 32 bits of the shared-aperture generic address = LDS byte offset
// (ISA 10.2: LDS aperture)
__device__ __forceinline__ unsigned lds_off_u32(const void* p) {
    return (unsigned)(unsigned long long)p;
}

// async global -> LDS, 16 bytes per lane (ASYNCcnt-tracked)
__device__ __forceinline__ void async_load_b128(unsigned lds_byte_off, const void* gaddr) {
    asm volatile("global_load_async_to_lds_b128 %0, %1, off"
                 :: "v"(lds_byte_off), "v"(gaddr)
                 : "memory");
}

__device__ __forceinline__ void wait_asynccnt_0() {
    asm volatile("s_wait_asynccnt 0x0" ::: "memory");
}
// allow the 12 most recent per-wave async ops to remain in flight
__device__ __forceinline__ void wait_asynccnt_12() {
    asm volatile("s_wait_asynccnt 0xc" ::: "memory");
}

// --- WMMA fragment loaders (wave32), per CDNA5 ISA 7.12.2 ------
// A matrix 16x32 f16: lane L: row = L&15, half = L>>4.
//   VGPR v in 0..3: K = 2v + 8*half ;  v in 4..7: K = 16 + 2(v-4) + 8*half.
__device__ __forceinline__ v16h load_a_frag(const _Float16* __restrict__ src, int ld) {
    const int lane = threadIdx.x & 31;
    const int row  = lane & 15;
    const int half = lane >> 4;
    v16h a;
#pragma unroll
    for (int v = 0; v < 8; ++v) {
        const int k = ((v < 4) ? (2 * v) : (16 + 2 * (v - 4))) + 8 * half;
        const h2 val = *(const h2*)(src + row * ld + k);
        a[2 * v]     = val.x;
        a[2 * v + 1] = val.y;
    }
    return a;
}

// B matrix 32x16 f16 (K x N), element (k,n) at src[n*ld + k]
// lane L: col = L&15, half = L>>4 ; VGPR v in 0..3: K = 2v + 8*half ; v in 4..7: K = 16 + 2(v-4) + 8*half.
__device__ __forceinline__ v16h load_b_frag(const _Float16* __restrict__ src, int ld) {
    const int lane = threadIdx.x & 31;
    const int col  = lane & 15;
    const int half = lane >> 4;
    v16h b;
#pragma unroll
    for (int v = 0; v < 8; ++v) {
        const int k = ((v < 4) ? (2 * v) : (16 + 2 * (v - 4))) + 8 * half;
        const h2 val = *(const h2*)(src + col * ld + k);
        b[2 * v]     = val.x;
        b[2 * v + 1] = val.y;
    }
    return b;
}

// 16x16 output tile, contraction over K (multiple of 32)
__device__ __forceinline__ v8f gemm_16x16(const _Float16* __restrict__ Asrc, int lda,
                                          const _Float16* __restrict__ Bsrc, int ldb,
                                          int K, v8f acc) {
    for (int k0 = 0; k0 < K; k0 += 32) {
        v16h a = load_a_frag(Asrc + k0, lda);
        v16h b = load_b_frag(Bsrc + k0, ldb);
        acc = __builtin_amdgcn_wmma_f32_16x16x32_f16(false, a, false, b,
                                                     (short)0, acc, false, false);
    }
    return acc;
}

// --------------------------------------------------------------------------
// k0: convert 6 weight matrices fp32 -> f16 into ws slots
__global__ void __launch_bounds__(256)
k0_cvt_weights(const float* __restrict__ lw1, const float* __restrict__ lw2,
               const float* __restrict__ rw1, const float* __restrict__ rw2,
               const float* __restrict__ sw,  const float* __restrict__ cw,
               _Float16* __restrict__ W16) {
    const int i = blockIdx.x * 256 + threadIdx.x;          // 0..16383
    const float* srcs[6] = {lw1, lw2, rw1, rw2, sw, cw};
    const float* s = srcs[blockIdx.y];
    W16[(size_t)blockIdx.y * (CDIM * CDIM) + i] = (_Float16)s[i];
}

// --------------------------------------------------------------------------
// k1: 16 positions per block, 128 threads (4 waves).
__global__ void __launch_bounds__(128)
k1_edges(const float* __restrict__ X, const unsigned char* __restrict__ mask,
         const float* __restrict__ ln1_g, const float* __restrict__ ln1_b,
         const _Float16* __restrict__ W16,
         const float* __restrict__ lb1, const float* __restrict__ lb2,
         const float* __restrict__ rb1, const float* __restrict__ rb2,
         const float* __restrict__ sb,
         _Float16* __restrict__ At, _Float16* __restrict__ Bt,
         float* __restrict__ G) {
    __shared__ __align__(16) float    xs[16 * CDIM];
    __shared__ __align__(16) _Float16 hs[16 * CDIM];
    __shared__ __align__(16) _Float16 t1s[16 * CDIM];
    __shared__ __align__(16) _Float16 t2s[16 * CDIM];
    __shared__ float mu_s[16], rs_s[16];

    const int tid = threadIdx.x;
    const int p0  = blockIdx.x * 16;

    // TDM: DMA the contiguous 16x128 fp32 tile (8KB) into LDS. Wave 0 only.
    if (tid < 32) {
        const unsigned long long ga = (unsigned long long)(X + (size_t)p0 * CDIM);
        u32x4 g0;
        g0[0] = 1u;                                   // count=1, user mode
        g0[1] = lds_off_u32(xs);                      // lds_addr
        g0[2] = (unsigned)ga;                         // global_addr[31:0]
        g0[3] = (unsigned)(ga >> 32) | (2u << 30);    // global_addr[56:32] | type=2
        u32x8 g1;
        g1[0] = (2u << 16);                           // workgroup_mask=0, data_size=4B
        const unsigned td0 = 2048;                    // tensor_dim0 (dwords)
        g1[1] = (td0 & 0xFFFFu) << 16;                // tensor_dim0[15:0]  @ bits 63:48
        g1[2] = ((td0 >> 16) & 0xFFFFu) | (1u << 16); // tensor_dim0[31:16], tensor_dim1=1
        g1[3] = (2048u << 16);                        // tile_dim0 = 2048   @ bits 127:112
        g1[4] = 0;                                    // tile_dim1 = tile_dim2 = 0
        g1[5] = 2048u;                                // tensor_dim0_stride low
        g1[6] = 0;
        g1[7] = 0;
        asm volatile("tensor_load_to_lds %0, %1" :: "s"(g0), "s"(g1) : "memory");
    }
    __builtin_amdgcn_s_wait_tensorcnt(0);
    __syncthreads();

    // LN stats: 8 threads per row, shuffle-reduce within groups of 8 lanes
    const int r = tid >> 3, g = tid & 7;
    {
        float s = 0.f, s2 = 0.f;
        for (int c = g * 16; c < g * 16 + 16; ++c) {
            float v = xs[r * CDIM + c];
            s += v; s2 += v * v;
        }
        s  += __shfl_xor(s, 1, 32);  s  += __shfl_xor(s, 2, 32);  s  += __shfl_xor(s, 4, 32);
        s2 += __shfl_xor(s2, 1, 32); s2 += __shfl_xor(s2, 2, 32); s2 += __shfl_xor(s2, 4, 32);
        if (g == 0) {
            float m  = s * (1.0f / CDIM);
            float var = s2 * (1.0f / CDIM) - m * m;
            mu_s[r] = m;
            rs_s[r] = rsqrtf(var + 1e-5f);
        }
    }
    __syncthreads();

    // normalized h in f16
    for (int c = g * 16; c < g * 16 + 16; ++c) {
        float h = (xs[r * CDIM + c] - mu_s[r]) * rs_s[r] * ln1_g[c] + ln1_b[c];
        hs[r * CDIM + c] = (_Float16)h;
    }
    __syncthreads();

    const int wave = tid >> 5;
    const int lane = tid & 31;
    const int nn   = lane & 15;
    const int half = lane >> 4;

    // first layer of both MLPs + gate (each wave: 2 column tiles of 16)
#pragma unroll
    for (int t = 0; t < 2; ++t) {
        const int ob = (wave * 2 + t) * 16;
        v8f a1 = {}; a1 = gemm_16x16(hs, CDIM, W16 + (size_t)MAT_LW1 * 16384 + ob * CDIM, CDIM, CDIM, a1);
        v8f a2 = {}; a2 = gemm_16x16(hs, CDIM, W16 + (size_t)MAT_RW1 * 16384 + ob * CDIM, CDIM, CDIM, a2);
        v8f ag = {}; ag = gemm_16x16(hs, CDIM, W16 + (size_t)MAT_SW  * 16384 + ob * CDIM, CDIM, CDIM, ag);
        const float b1 = lb1[ob + nn];
        const float b2 = rb1[ob + nn];
        const float bg = sb[ob + nn];
#pragma unroll
        for (int v = 0; v < 8; ++v) {
            const int m = v + 8 * half;
            t1s[m * CDIM + ob + nn] = (_Float16)sigmoidf_(a1[v] + b1);
            t2s[m * CDIM + ob + nn] = (_Float16)sigmoidf_(a2[v] + b2);
            G[(size_t)(p0 + m) * CDIM + ob + nn] = sigmoidf_(ag[v] + bg);
        }
    }
    __syncthreads();

    // second layer + mask, write channel-major f16 A,B (contiguous 8 f16/lane)
#pragma unroll
    for (int t = 0; t < 2; ++t) {
        const int ob = (wave * 2 + t) * 16;
        v8f aa = {}; aa = gemm_16x16(t1s, CDIM, W16 + (size_t)MAT_LW2 * 16384 + ob * CDIM, CDIM, CDIM, aa);
        v8f ab = {}; ab = gemm_16x16(t2s, CDIM, W16 + (size_t)MAT_RW2 * 16384 + ob * CDIM, CDIM, CDIM, ab);
        const float bA = lb2[ob + nn];
        const float bB = rb2[ob + nn];
        v8h oa, obv;
#pragma unroll
        for (int v = 0; v < 8; ++v) {
            const int m = v + 8 * half;
            const float mk = mask[p0 + m] ? 1.0f : 0.0f;   // bool, 1 byte/elem
            oa[v]  = (_Float16)((aa[v] + bA) * mk);
            obv[v] = (_Float16)((ab[v] + bB) * mk);
        }
        const size_t base = (size_t)(ob + nn) * NN + p0 + 8 * half;
        *(v8h*)(At + base) = oa;
        *(v8h*)(Bt + base) = obv;
    }
}

// --------------------------------------------------------------------------
// k2: triangle GEMM, double-buffered async-LDS staging.
// grid = (4 jt, 8 it, 128 c), block = 128 threads (4 waves).
// Block tile: 64 rows x 128 cols. Wave (mw,nw): 32x64 -> 8 accumulators.
// Stage = K-chunk of 64: A(64x64) + B(128x64) f16 per buffer (24KB x2 LDS).
// Each wave issues 12 async b128 ops per stage; s_wait_asynccnt 12 retires
// the previous stage while the next one streams in behind the WMMAs.
#define KSTEP 64
__global__ void __launch_bounds__(128)
k2_triangle(const _Float16* __restrict__ At, const _Float16* __restrict__ Bt,
            float* __restrict__ T) {
    __shared__ __align__(16) _Float16 As[2][64 * KSTEP];    // ld = KSTEP
    __shared__ __align__(16) _Float16 Bs[2][128 * KSTEP];   // ld = KSTEP

    const int c    = blockIdx.z;
    const int tid  = threadIdx.x;
    const int wave = tid >> 5;
    const int lane = tid & 31;
    const int nn   = lane & 15;
    const int half = lane >> 4;
    const int mw   = wave >> 1;         // 0..1 : row half of block tile
    const int nw   = wave & 1;          // 0..1 : col half of block tile

    const _Float16* __restrict__ A = At + (size_t)c * NN;
    const _Float16* __restrict__ B = Bt + (size_t)c * NN;
    const int i0 = blockIdx.y * 64;
    const int j0 = blockIdx.x * 128;

    const unsigned asOff[2] = {lds_off_u32(As[0]), lds_off_u32(As[1])};
    const unsigned bsOff[2] = {lds_off_u32(Bs[0]), lds_off_u32(Bs[1])};

    // issue one stage (per-thread: 4 A + 8 B async b128 = 12 per wave)
    auto stage = [&](int buf, int k0) {
#pragma unroll
        for (int q = tid; q < 512; q += 128) {            // A: 64 rows x 128B
            const int row = q >> 3, c16 = (q & 7) * 16;
            async_load_b128(asOff[buf] + row * (KSTEP * 2) + c16,
                            (const char*)(A + ((size_t)(i0 + row) * NPOS + k0)) + c16);
        }
#pragma unroll
        for (int q = tid; q < 1024; q += 128) {           // B: 128 rows x 128B
            const int row = q >> 3, c16 = (q & 7) * 16;
            async_load_b128(bsOff[buf] + row * (KSTEP * 2) + c16,
                            (const char*)(B + ((size_t)(j0 + row) * NPOS + k0)) + c16);
        }
    };

    v8f acc[2][4] = {};
    stage(0, 0);
    const int NSTAGE = NPOS / KSTEP;    // 8
    for (int s = 0; s < NSTAGE; ++s) {
        const int cur = s & 1;
        if (s + 1 < NSTAGE) {
            stage(1 - cur, (s + 1) * KSTEP);  // buf (1-cur) last read 2 stages ago
            wait_asynccnt_12();               // oldest stage (cur) complete, in-order
        } else {
            wait_asynccnt_0();
        }
        __syncthreads();                      // all waves' DMA for cur complete

        const _Float16* as = As[cur];
        const _Float16* bs = Bs[cur];
#pragma unroll
        for (int kk = 0; kk < KSTEP; kk += 32) {
            v16h af0 = load_a_frag(as + (mw * 32 + 0) * KSTEP + kk, KSTEP);
            v16h af1 = load_a_frag(as + (mw * 32 + 16) * KSTEP + kk, KSTEP);
#pragma unroll
            for (int t = 0; t < 4; ++t) {
                v16h bf = load_b_frag(bs + (nw * 64 + t * 16) * KSTEP + kk, KSTEP);
                acc[0][t] = __builtin_amdgcn_wmma_f32_16x16x32_f16(false, af0, false, bf,
                                                                   (short)0, acc[0][t], false, false);
                acc[1][t] = __builtin_amdgcn_wmma_f32_16x16x32_f16(false, af1, false, bf,
                                                                   (short)0, acc[1][t], false, false);
            }
        }
        __syncthreads();   // done reading cur before it is overwritten at s+2
    }

#pragma unroll
    for (int x = 0; x < 2; ++x) {
#pragma unroll
        for (int t = 0; t < 4; ++t) {
#pragma unroll
            for (int v = 0; v < 8; ++v) {
                const int i = i0 + mw * 32 + x * 16 + v + 8 * half;
                const int j = j0 + nw * 64 + t * 16 + nn;
                T[((size_t)i * NPOS + j) * CDIM + c] = acc[x][t][v];
            }
        }
    }
}

// --------------------------------------------------------------------------
// k3: LN2 + output linear * gate, in place on d_out (T -> Y).
__global__ void __launch_bounds__(128)
k3_out(float* __restrict__ TY,
       const float* __restrict__ ln2_g, const float* __restrict__ ln2_b,
       const _Float16* __restrict__ W16, const float* __restrict__ cb,
       const float* __restrict__ G) {
    __shared__ __align__(16) float    xs[16 * CDIM];
    __shared__ __align__(16) _Float16 hs[16 * CDIM];
    __shared__ float mu_s[16], rs_s[16];

    const int tid = threadIdx.x;
    const int p0  = blockIdx.x * 16;

    {
        const float4* src = (const float4*)(TY + (size_t)p0 * CDIM);
        float4* dst = (float4*)xs;
        for (int i = tid; i < (16 * CDIM) / 4; i += 128) dst[i] = src[i];
    }
    __syncthreads();

    const int r = tid >> 3, g = tid & 7;
    {
        float s = 0.f, s2 = 0.f;
        for (int c = g * 16; c < g * 16 + 16; ++c) {
            float v = xs[r * CDIM + c];
            s += v; s2 += v * v;
        }
        s  += __shfl_xor(s, 1, 32);  s  += __shfl_xor(s, 2, 32);  s  += __shfl_xor(s, 4, 32);
        s2 += __shfl_xor(s2, 1, 32); s2 += __shfl_xor(s2, 2, 32); s2 += __shfl_xor(s2, 4, 32);
        if (g == 0) {
            float m  = s * (1.0f / CDIM);
            float var = s2 * (1.0f / CDIM) - m * m;
            mu_s[r] = m;
            rs_s[r] = rsqrtf(var + 1e-5f);
        }
    }
    __syncthreads();

    for (int c = g * 16; c < g * 16 + 16; ++c) {
        float h = (xs[r * CDIM + c] - mu_s[r]) * rs_s[r] * ln2_g[c] + ln2_b[c];
        hs[r * CDIM + c] = (_Float16)h;
    }
    __syncthreads();

    const int wave = tid >> 5;
    const int lane = tid & 31;
    const int nn   = lane & 15;
    const int half = lane >> 4;

#pragma unroll
    for (int t = 0; t < 2; ++t) {
        const int ob = (wave * 2 + t) * 16;
        v8f acc = {};
        acc = gemm_16x16(hs, CDIM, W16 + (size_t)MAT_CW * 16384 + ob * CDIM, CDIM, CDIM, acc);
        const float bc = cb[ob + nn];
#pragma unroll
        for (int v = 0; v < 8; ++v) {
            const int m = v + 8 * half;
            const size_t idx = (size_t)(p0 + m) * CDIM + ob + nn;
            TY[idx] = (acc[v] + bc) * G[idx];
        }
    }
}

// --------------------------------------------------------------------------
extern "C" void kernel_launch(void* const* d_in, const int* in_sizes, int n_in,
                              void* d_out, int out_size, void* d_ws, size_t ws_size,
                              hipStream_t stream) {
    const float* X            = (const float*)d_in[0];
    const unsigned char* mask = (const unsigned char*)d_in[1];  // jax bool = 1 byte
    const float* ln1_g = (const float*)d_in[2];
    const float* ln1_b = (const float*)d_in[3];
    const float* lw1   = (const float*)d_in[4];
    const float* lb1   = (const float*)d_in[5];
    const float* lw2   = (const float*)d_in[6];
    const float* lb2   = (const float*)d_in[7];
    const float* rw1   = (const float*)d_in[8];
    const float* rb1   = (const float*)d_in[9];
    const float* rw2   = (const float*)d_in[10];
    const float* rb2   = (const float*)d_in[11];
    const float* sw    = (const float*)d_in[12];
    const float* sb    = (const float*)d_in[13];
    const float* ln2_g = (const float*)d_in[14];
    const float* ln2_b = (const float*)d_in[15];
    const float* cw    = (const float*)d_in[16];
    const float* cb    = (const float*)d_in[17];

    char* ws = (char*)d_ws;
    _Float16* W16 = (_Float16*)(ws + OFF_W16);
    _Float16* At  = (_Float16*)(ws + OFF_AT);
    _Float16* Bt  = (_Float16*)(ws + OFF_BT);
    float*    G   = (float*)(ws + OFF_G);
    float*    TY  = (float*)d_out;   // reused: T scratch then final Y (in place)

    k0_cvt_weights<<<dim3(64, 6), 256, 0, stream>>>(lw1, lw2, rw1, rw2, sw, cw, W16);
    k1_edges<<<dim3(NN / 16), 128, 0, stream>>>(X, mask, ln1_g, ln1_b, W16,
                                                lb1, lb2, rb1, rb2, sb, At, Bt, G);
    k2_triangle<<<dim3(4, 8, CDIM), 128, 0, stream>>>(At, Bt, TY);
    k3_out<<<dim3(NN / 16), 128, 0, stream>>>(TY, ln2_g, ln2_b, W16, cb, G);
}
